// kn_mlp_Conv1D_34514357190887
// MI455X (gfx1250) — matrix operand order repaired and checked by reading, not compiled, not run on hardware
//
#include <hip/hip_runtime.h>
#include <math.h>

// Problem constants (fixed by the reference)
#define R_  8
#define M1_ 32
#define N1_ 64
#define M2_ 24
#define N2_ 48
#define DIN 768      // M1*M2
#define BS_ 4096     // B*S
#define TOK_PER_WG 8
#define NTHREADS 256

// Padded LDS row strides (elements): 264*2B=528B and 520*2B=1040B are ≡ 4 banks
// (mod 64) between consecutive fragment lanes -> conflict-free ds_load_b128.
#define LDA1 264     // A1s rows (logical 256)
#define LDG2 264     // G2  rows (logical 256)
#define LDA2 520     // A2s rows (logical 512)
#define LDFT 520     // FT  rows (logical 512)

typedef __attribute__((ext_vector_type(16))) __bf16       v16bf;
typedef __attribute__((ext_vector_type(2)))  __bf16       v2bf;
typedef __attribute__((ext_vector_type(8)))  float        v8f;
typedef __attribute__((ext_vector_type(4)))  unsigned int u32x4;

union FragBF { v16bf v; u32x4 u[2]; };

// f32 -> bf16 (RNE) via native casts; vector-pair form fuses to one
// two-operand v_cvt_pk_bf16_f32.
__device__ __forceinline__ unsigned pk2(float a, float b) {
  v2bf p;
  p.x = (__bf16)a;
  p.y = (__bf16)b;
  return __builtin_bit_cast(unsigned, p);
}
__device__ __forceinline__ unsigned short f2bf(float f) {
  __bf16 h = (__bf16)f;
  return __builtin_bit_cast(unsigned short, h);
}
// exact floor(c/24) for c < 768 in two VALU ops
__device__ __forceinline__ int div24(int c) { return (c * 2731) >> 16; }

// A operand (16x32 bf16): lane<16 row M=lane, K = {k0..k0+7, k0+16..k0+23};
// lane>=16 row M=lane-16, K = {k0+8..15, k0+24..31}. base is row-major [rows][ldk].
__device__ __forceinline__ FragBF lds_frag_a(const unsigned short* base, int ldk,
                                             int row0, int k0, int lane) {
  int r  = row0 + (lane & 15);
  int hi = (lane >> 4) & 1;
  const unsigned short* p = base + r * ldk + k0 + hi * 8;
  FragBF f;
  f.u[0] = *(const u32x4*)(p);
  f.u[1] = *(const u32x4*)(p + 16);
  return f;
}
// B operand (32x16 bf16): lane<16 col N=lane holds K = {k0..k0+7, k0+16..k0+23}; lane>=16 holds
// K = {k0+8..15, k0+24..31}. base stores B column-major: [cols][ldk] with K contiguous.
__device__ __forceinline__ FragBF lds_frag_b(const unsigned short* base, int ldk,
                                             int col0, int k0, int lane) {
  int c  = col0 + (lane & 15);
  int hi = (lane >> 4) & 1;
  const unsigned short* p = base + c * ldk + k0 + hi * 8;
  FragBF f;
  f.u[0] = *(const u32x4*)(p);
  f.u[1] = *(const u32x4*)(p + 16);
  return f;
}
__device__ __forceinline__ v8f wmma_bf16(FragBF a, FragBF b, v8f c) {
  return __builtin_amdgcn_wmma_f32_16x16x32_bf16(false, a.v, false, b.v,
                                                 (short)0, c, false, false);
}

__global__ __launch_bounds__(NTHREADS) void kron_mlp_fused(
    const float* __restrict__ x,  const float* __restrict__ A1,
    const float* __restrict__ B1, const float* __restrict__ A2,
    const float* __restrict__ B2, const float* __restrict__ b1,
    const float* __restrict__ b2, const int* __restrict__ p_inv,
    const int* __restrict__ q,    float* __restrict__ out)
{
  // Weights (bf16, resident for all tokens of this WG)
  __shared__ __align__(16) unsigned short A1s[N1_ * LDA1];         // [64][264] row-major (A op)
  __shared__ __align__(16) unsigned short B1sT[(R_ * N2_) * 32];   // [384][32] col-major, K pad 24->32
  __shared__ __align__(16) unsigned short A2s[M1_ * LDA2];         // [32][520] row-major (A op)
  __shared__ __align__(16) unsigned short B2sT[(R_ * M2_) * 64];   // [192][64] col-major, K pad 48->64
  // Per-token staging
  __shared__ __align__(16) unsigned short Xs[32 * 32];             // [32][32] K pad 24->32
  __shared__ __align__(16) unsigned short Hs[N1_ * 64];            // [64][64] K pad 48->64
  // G2 (S1 out / S2 B-op) and FT (S3 out / S4 B-op) are time-disjoint -> overlay.
  // (G2 tail overlaps FT row 24; that row only feeds discarded S4 columns.)
  __shared__ __align__(16) union {
    unsigned short G2[N2_ * LDG2];   // [48][264]  G'^T: [j][r*32+m]
    unsigned short FT[32 * LDFT];    // [32][520]  F'^T: [m2][r*64+n1]
  } GF;
  __shared__ __align__(16) float outb[DIN];

  const int tid  = threadIdx.x;
  const int lane = tid & 31;
  // wave id is uniform per wave: tell the compiler so tile loops become scalar
  const int wid  = __builtin_amdgcn_readfirstlane(tid >> 5);

  // ---------------- weight preload (f32 -> bf16) ----------------
  for (int i = tid; i < N1_ * 256; i += NTHREADS) {            // A1s[n][r*32+m]=A1[r][n][m]
    int n = i >> 8, c = i & 255, r = c >> 5, m = c & 31;
    A1s[n * LDA1 + c] = f2bf(A1[r * (N1_ * M1_) + n * M1_ + m]);
  }
  for (int i = tid; i < (R_ * N2_) * 32; i += NTHREADS) {      // B1sT[r*48+j][k]=B1[r][k][j]
    int cidx = i >> 5, k = i & 31;
    int r = cidx / N2_, j = cidx - r * N2_;
    B1sT[i] = (k < M2_) ? f2bf(B1[r * (M2_ * N2_) + k * N2_ + j]) : (unsigned short)0;
  }
  for (int i = tid; i < M1_ * 512; i += NTHREADS) {            // A2s[m1][r*64+n]=A2[r][m1][n]
    int m = i >> 9, c = i & 511, r = c >> 6, n = c & 63;
    A2s[m * LDA2 + c] = f2bf(A2[r * (M1_ * N1_) + m * N1_ + n]);
  }
  for (int i = tid; i < (R_ * M2_) * 64; i += NTHREADS) {      // B2sT[r*24+m2][k]=B2[r][k][m2]
    int cidx = i >> 6, k = i & 63;
    int r = div24(cidx), m2 = cidx - r * M2_;
    B2sT[i] = (k < N2_) ? f2bf(B2[r * (N2_ * M2_) + k * M2_ + m2]) : (unsigned short)0;
  }
  for (int i = tid; i < 32 * 32; i += NTHREADS) Xs[i] = 0;          // K-pad cols stay 0
  for (int i = tid; i < N1_ * 64; i += NTHREADS) Hs[i] = 0;         // K-pad cols stay 0
  for (int i = tid; i < 32 * LDFT; i += NTHREADS) GF.FT[i] = 0;
  __syncthreads();

  const int token0 = blockIdx.x * TOK_PER_WG;
  for (int it = 0; it < TOK_PER_WG; ++it) {
    const int token = token0 + it;
    const float* xrow = x + (size_t)token * DIN;

    // prefetch next token's row while we compute this one (global_prefetch_b8)
    if (it + 1 < TOK_PER_WG) {
      __builtin_prefetch(xrow + DIN + tid * 3, 0, 0);   // 256 thr * 12B covers 3KB row
    }

    // permuted gather: X[m1][m2] = x[token][p_inv[m1*24+m2]]
    for (int i = tid; i < DIN; i += NTHREADS) {
      int m1 = div24(i), m2 = i - m1 * M2_;
      Xs[m1 * 32 + m2] = f2bf(xrow[p_inv[i]]);
    }
    __syncthreads();

    // S1: G[32][384] = X @ B1stack (K=24 pad 32); scatter to G2[j][r*32+m] (bf16)
    for (int tt = wid; tt < 2 * 24; tt += 8) {
      int mt = tt / 24, nt = tt - mt * 24;
      FragBF a = lds_frag_a(Xs, 32, mt * 16, 0, lane);
      FragBF b = lds_frag_b(B1sT, 32, nt * 16, 0, lane);
      v8f acc = {0.f, 0.f, 0.f, 0.f, 0.f, 0.f, 0.f, 0.f};
      acc = wmma_bf16(a, b, acc);
      // col = nt*16 + (lane&15); 48 = 3*16 so r is tile-uniform (scalar)
      int r = nt / 3;                              // scalar
      int j = (nt - r * 3) * 16 + (lane & 15);     // 0..47
      int m0 = mt * 16 + ((lane >> 4) << 3);       // 8 consecutive rows per lane
      u32x4 w;
      w.x = pk2(acc[0], acc[1]); w.y = pk2(acc[2], acc[3]);
      w.z = pk2(acc[4], acc[5]); w.w = pk2(acc[6], acc[7]);
      *(u32x4*)(&GF.G2[j * LDG2 + r * M1_ + m0]) = w;
    }
    __syncthreads();

    // S2: h[64][48] = A1stack[64,256] @ G'[256,48]; +b1; exact GELU; -> Hs bf16
    for (int tt = wid; tt < 4 * 3; tt += 8) {
      int mt = tt / 3, nt = tt - mt * 3;
      v8f acc = {0.f, 0.f, 0.f, 0.f, 0.f, 0.f, 0.f, 0.f};
#pragma unroll
      for (int ks = 0; ks < 8; ++ks) {
        FragBF a = lds_frag_a(A1s, LDA1, mt * 16, ks * 32, lane);
        FragBF b = lds_frag_b(GF.G2, LDG2, nt * 16, ks * 32, lane);
        acc = wmma_bf16(a, b, acc);
      }
      int j  = nt * 16 + (lane & 15);
      int n0 = mt * 16 + ((lane >> 4) << 3);
#pragma unroll
      for (int v = 0; v < 8; ++v) {
        int n1 = n0 + v;
        float h = acc[v] + b1[n1 * N2_ + j];
        h = 0.5f * h * (1.0f + erff(h * 0.70710678118654752440f));   // exact gelu
        Hs[n1 * 64 + j] = f2bf(h);
      }
    }
    __syncthreads();

    // S3: F[64][192] = H @ B2stack (K=48 pad 64); scatter to FT[m2][r*64+n1]
    for (int tt = wid; tt < 4 * 12; tt += 8) {
      int mt = tt / 12, nt = tt - mt * 12;
      v8f acc = {0.f, 0.f, 0.f, 0.f, 0.f, 0.f, 0.f, 0.f};
#pragma unroll
      for (int ks = 0; ks < 2; ++ks) {
        FragBF a = lds_frag_a(Hs, 64, mt * 16, ks * 32, lane);
        FragBF b = lds_frag_b(B2sT, 64, nt * 16, ks * 32, lane);
        acc = wmma_bf16(a, b, acc);
      }
      int col = nt * 16 + (lane & 15);           // 0..191
      int r = div24(col), m2 = col - r * M2_;
      int n0 = mt * 16 + ((lane >> 4) << 3);
      u32x4 w;
      w.x = pk2(acc[0], acc[1]); w.y = pk2(acc[2], acc[3]);
      w.z = pk2(acc[4], acc[5]); w.w = pk2(acc[6], acc[7]);
      *(u32x4*)(&GF.FT[m2 * LDFT + r * N1_ + n0]) = w;
    }
    __syncthreads();

    // S4: out[32][24] = A2stack[32,512] @ F'[512,24]; +b2 -> outb (f32)
    for (int tt = wid; tt < 4; tt += 8) {        // waves 0..3
      int mt = tt >> 1, nt = tt & 1;
      v8f acc = {0.f, 0.f, 0.f, 0.f, 0.f, 0.f, 0.f, 0.f};
#pragma unroll
      for (int ks = 0; ks < 16; ++ks) {
        FragBF a = lds_frag_a(A2s, LDA2, mt * 16, ks * 32, lane);
        FragBF b = lds_frag_b(GF.FT, LDFT, nt * 16, ks * 32, lane);
        acc = wmma_bf16(a, b, acc);
      }
      int col = nt * 16 + (lane & 15);           // valid < 24
      int m0  = mt * 16 + ((lane >> 4) << 3);
      if (col < M2_) {
#pragma unroll
        for (int v = 0; v < 8; ++v) {
          int m1i = m0 + v;
          outb[m1i * M2_ + col] = acc[v] + b2[m1i * M2_ + col];
        }
      }
    }
    __syncthreads();

    // output permutation: out[token][i] = outb[q[i]]
    float* orow = out + (size_t)token * DIN;
    for (int i = tid; i < DIN; i += NTHREADS) orow[i] = outb[q[i]];
    // no trailing barrier needed: next write to outb is after >=4 barriers
  }
}

extern "C" void kernel_launch(void* const* d_in, const int* in_sizes, int n_in,
                              void* d_out, int out_size, void* d_ws, size_t ws_size,
                              hipStream_t stream) {
  (void)in_sizes; (void)n_in; (void)out_size; (void)d_ws; (void)ws_size;
  const float* x  = (const float*)d_in[0];
  const float* A1 = (const float*)d_in[1];
  const float* B1 = (const float*)d_in[2];
  const float* A2 = (const float*)d_in[3];
  const float* B2 = (const float*)d_in[4];
  const float* b1 = (const float*)d_in[5];
  const float* b2 = (const float*)d_in[6];
  const int* p_inv = (const int*)d_in[7];
  const int* qi    = (const int*)d_in[8];
  float* out = (float*)d_out;

  dim3 grid(BS_ / TOK_PER_WG);   // 512 workgroups
  dim3 block(NTHREADS);          // 8 wave32s
  kron_mlp_fused<<<grid, block, 0, stream>>>(x, A1, B1, A2, B2, b1, b2, p_inv, qi, out);
}
